// VMFPointNetDensityEncoder_69020124446966
// MI455X (gfx1250) — compile-verified, matrix operand order checked by reading, not run on hardware
//
#include <hip/hip_runtime.h>
#include <hip/hip_bf16.h>
#include <math.h>

// ---------------------------------------------------------------------------
// VMF PointNet density encoder for MI455X (gfx1250), wave32 + WMMA f16 +
// async global->LDS tile staging (ASYNCcnt).
// ---------------------------------------------------------------------------

typedef _Float16 f16;
typedef __attribute__((ext_vector_type(16))) _Float16 v16h;
typedef __attribute__((ext_vector_type(8)))  _Float16 v8h;
typedef __attribute__((ext_vector_type(8)))  float    v8f;

constexpr int kB = 8;
constexpr int kN = 2048;
constexpr float kEps = 1e-5f;

// ---------------- workspace layout (bytes, all 64-aligned) ----------------
constexpr long long OFF_H    = 0;                       // f32 [8][512][2048]
constexpr long long OFF_T    = OFF_H    + 33554432LL;   // f32 [8][512][2048] (xr / conv out)
constexpr long long OFF_HB   = OFF_T    + 33554432LL;   // f16 [8][C][2048]  (GEMM A)
constexpr long long OFF_HNC  = OFF_HB   + 16777216LL;   // f16 [8][2048][C]  (GEMM B, K-contig)
constexpr long long OFF_WT   = OFF_HNC  + 16777216LL;   // f16 [8][2048(m)][2048(n)] attn^T
constexpr long long OFF_WF   = OFF_WT   + 67108864LL;   // f16 weights pool
constexpr long long OFF_XN   = OFF_WF   + 1081600LL;    // f32 [8][2048][3]
constexpr long long OFF_DINV = OFF_XN   + 196608LL;     // f32 [8][2048]
constexpr long long OFF_DENS = OFF_DINV + 65536LL;      // f32 [8][2048]
constexpr long long OFF_BMAX = OFF_DENS + 65536LL;      // f32 [8]
constexpr long long OFF_S    = OFF_BMAX + 64LL;         // f32 scalar (||wqk||^2)
constexpr long long OFF_R    = OFF_S    + 64LL;         // f32 [8][2048] row denom
constexpr long long OFF_CS   = OFF_R    + 65536LL;      // f32 [8][2048] col sum
constexpr long long OFF_BN   = OFF_CS   + 65536LL;      // f32 [512][2] bn stats
constexpr long long OFF_G    = OFF_BN   + 4096LL;       // f32 [8][512] max pool
// total ~161.5 MiB

// f16 weight sub-offsets (elements)
constexpr long long WF_W2  = 0;        // 128x128
constexpr long long WF_W3  = 16384;    // 256x128
constexpr long long WF_W4  = 49152;    // 512x256
constexpr long long WF_AT1 = 180224;   // 128x128
constexpr long long WF_AT2 = 196608;   // 128x128
constexpr long long WF_AT3 = 212992;   // 256x256
constexpr long long WF_AT4 = 278528;   // 512x512

// ---- gfx1250 async global->LDS copy (16 B per lane), tracked by ASYNCcnt ---
__device__ __forceinline__ void async_load_b128(unsigned lds_off,
                                                unsigned long long gaddr)
{
    asm volatile("global_load_async_to_lds_b128 %0, %1, off"
                 :: "v"(lds_off), "v"(gaddr) : "memory");
}

__device__ __forceinline__ void wait_async0()
{
#if __has_builtin(__builtin_amdgcn_s_wait_asynccnt)
    __builtin_amdgcn_s_wait_asynccnt(0);
#else
    asm volatile("s_wait_asynccnt 0x0" ::: "memory");
#endif
}

// ---------------------------------------------------------------------------
// LDS-tiled WMMA GEMM:
//   C[b][m][nn] (f32) = sum_k A[b][m][k] * Bt[b][nn][k]   (+bias[m])
// A, Bt f16 row-major, leading dim == K (K-contiguous).
// Block tile 64(M) x 64(N) x 32(K), double-buffered async LDS staging.
// 8 waves in 4(M) x 2(N); each wave: one A fragment -> two 16x16 WMMA tiles.
// grid = (M/64, Nn/64, batch), block = 256.
// ---------------------------------------------------------------------------
constexpr int kPadHalf = 40;          // 32 f16 data + 8 f16 pad (80 B row, bank-safe)
constexpr unsigned kTileBytes = 64u * kPadHalf * sizeof(f16);   // 5120 B per buffer

__global__ __launch_bounds__(256)
void k_gemm_wmma(const f16* __restrict__ A, const f16* __restrict__ Bt,
                 float* __restrict__ C, const float* __restrict__ bias,
                 int Nn, int K,
                 long long strideA, long long strideB, long long strideC)
{
    __shared__ __align__(16) f16 smA[2][64][kPadHalf];
    __shared__ __align__(16) f16 smB[2][64][kPadHalf];

    const int b = blockIdx.z;
    A  += (long long)b * strideA;     // strideA == 0 => batch-shared weights
    Bt += (long long)b * strideB;
    C  += (long long)b * strideC;

    const int tid  = threadIdx.x;
    const int wave = tid >> 5;
    const int lane = tid & 31;
    const int half = lane >> 4;
    const int r    = lane & 15;

    const int mt0 = blockIdx.x * 64;
    const int nt0 = blockIdx.y * 64;

    // ---- loader mapping: 256 lanes x 16 B cover one 64x64-byte tile ----
    const int lrow = tid >> 2;        // 0..63
    const int lseg = tid & 3;         // 4 x 16 B segments per 64 B row
    const f16* gA = A  + (long long)(mt0 + lrow) * K + lseg * 8;
    const f16* gB = Bt + (long long)(nt0 + lrow) * K + lseg * 8;
    const unsigned ldsA0 = (unsigned)(unsigned long long)&smA[0][lrow][lseg * 8];
    const unsigned ldsB0 = (unsigned)(unsigned long long)&smB[0][lrow][lseg * 8];

    // prologue: stage K-tile 0 into buffer 0
    async_load_b128(ldsA0, (unsigned long long)gA);
    async_load_b128(ldsB0, (unsigned long long)gB);

    const int mrow = (wave & 3) * 16 + r;       // A row inside tile
    const int nc0  = (wave >> 2) * 32;          // N sub-base inside tile
    const int kb   = half * 8;                  // ISA 16-bit fragment K split

    v8f acc0 = {}, acc1 = {};
    int buf = 0;
    for (int k0 = 0; k0 < K; k0 += 32) {
        wait_async0();                // my staged tile is in LDS
        __syncthreads();              // everyone's tile is in LDS
        if (k0 + 32 < K) {            // stage next K-tile into other buffer
            const unsigned nb = (unsigned)(buf ^ 1) * kTileBytes;
            async_load_b128(ldsA0 + nb, (unsigned long long)(gA + k0 + 32));
            async_load_b128(ldsB0 + nb, (unsigned long long)(gB + k0 + 32));
        }
        // ---- fragments from LDS (ds_load_b128) ----
        const f16* Ar  = &smA[buf][mrow][0];
        const f16* Br0 = &smB[buf][nc0 + r][0];
        const f16* Br1 = &smB[buf][nc0 + 16 + r][0];
        v8h alo = *(const v8h*)(Ar + kb);
        v8h ahi = *(const v8h*)(Ar + kb + 16);
        v8h b0l = *(const v8h*)(Br0 + kb);
        v8h b0h = *(const v8h*)(Br0 + kb + 16);
        v8h b1l = *(const v8h*)(Br1 + kb);
        v8h b1h = *(const v8h*)(Br1 + kb + 16);
        v16h af, bf0, bf1;
        #pragma unroll
        for (int j = 0; j < 8; ++j) {
            af[j]  = alo[j]; af[j + 8]  = ahi[j];
            bf0[j] = b0l[j]; bf0[j + 8] = b0h[j];
            bf1[j] = b1l[j]; bf1[j + 8] = b1h[j];
        }
        acc0 = __builtin_amdgcn_wmma_f32_16x16x32_f16(
                   false, af, false, bf0, (short)0, acc0, false, false);
        acc1 = __builtin_amdgcn_wmma_f32_16x16x32_f16(
                   false, af, false, bf1, (short)0, acc1, false, false);
        buf ^= 1;
    }

    // D layout: VGPR ridx -> M = ridx + 8*half inside tile, N = lane&15
    #pragma unroll
    for (int ridx = 0; ridx < 8; ++ridx) {
        const int mm = mt0 + (wave & 3) * 16 + ridx + 8 * half;
        float v0 = acc0[ridx], v1 = acc1[ridx];
        if (bias) { v0 += bias[mm]; v1 += bias[mm]; }
        C[(long long)mm * Nn + nt0 + nc0 + r]      = v0;
        C[(long long)mm * Nn + nt0 + nc0 + 16 + r] = v1;
    }
}

// ---------------------------------------------------------------------------
// Density pipeline
// ---------------------------------------------------------------------------
__global__ void k_normalize(const float* __restrict__ x, float* __restrict__ xn)
{
    int i = blockIdx.x * blockDim.x + threadIdx.x;   // over B*N points
    if (i >= kB * kN) return;
    float a = x[3 * i], b = x[3 * i + 1], c = x[3 * i + 2];
    float inv = rsqrtf(a * a + b * b + c * c);
    xn[3 * i] = a * inv; xn[3 * i + 1] = b * inv; xn[3 * i + 2] = c * inv;
}

// one block per row: dist to all m in LDS, then 32x (argmin + mask)
__global__ __launch_bounds__(256)
void k_density(const float* __restrict__ xn, float* __restrict__ dinv)
{
    __shared__ float dist[kN];
    __shared__ float rv[256];
    __shared__ int   ri[256];
    const int row = blockIdx.x;          // b*kN + n
    const int b   = row / kN;
    const int tid = threadIdx.x;
    const float x0 = xn[3 * row], y0 = xn[3 * row + 1], z0 = xn[3 * row + 2];
    const float* base = xn + (long long)b * kN * 3;
    for (int m = tid; m < kN; m += 256) {
        dist[m] = 1.0f - (x0 * base[3 * m] + y0 * base[3 * m + 1] + z0 * base[3 * m + 2]);
    }
    __syncthreads();
    __shared__ float acc_s;
    if (tid == 0) acc_s = 0.0f;
    for (int it = 0; it < 32; ++it) {
        float best = 3.0e38f; int bi = 0;
        for (int m = tid; m < kN; m += 256) {
            float d = dist[m];
            if (d < best) { best = d; bi = m; }
        }
        rv[tid] = best; ri[tid] = bi;
        __syncthreads();
        for (int s = 128; s > 0; s >>= 1) {
            if (tid < s && rv[tid + s] < rv[tid]) { rv[tid] = rv[tid + s]; ri[tid] = ri[tid + s]; }
            __syncthreads();
        }
        if (tid == 0) { acc_s += rv[0]; dist[ri[0]] = 3.0e38f; }
        __syncthreads();
    }
    if (tid == 0) dinv[row] = 32.0f / acc_s;   // 1 / mean(top-32 smallest)
}

__global__ __launch_bounds__(256)
void k_batchmax(const float* __restrict__ dinv, float* __restrict__ bmax)
{
    __shared__ float rv[256];
    const int b = blockIdx.x, tid = threadIdx.x;
    float m = -3.0e38f;
    for (int n = tid; n < kN; n += 256) m = fmaxf(m, dinv[b * kN + n]);
    rv[tid] = m; __syncthreads();
    for (int s = 128; s > 0; s >>= 1) {
        if (tid < s) rv[tid] = fmaxf(rv[tid], rv[tid + s]);
        __syncthreads();
    }
    if (tid == 0) bmax[b] = rv[0];
}

__global__ void k_dnorm(const float* __restrict__ dinv, const float* __restrict__ bmax,
                        float* __restrict__ dens)
{
    int i = blockIdx.x * blockDim.x + threadIdx.x;
    if (i >= kB * kN) return;
    float g = bmax[0];                 // global max (jnp.max over all of inv)
    #pragma unroll
    for (int b = 1; b < kB; ++b) g = fmaxf(g, bmax[b]);
    dens[i] = dinv[i] / g;
}

// ---------------------------------------------------------------------------
// Attention statistics (energy is rank-1: s * d[n] * d[m])
// ---------------------------------------------------------------------------
__global__ void k_sumsq(const float* __restrict__ wqk, int D, float* __restrict__ s)
{
    if (blockIdx.x == 0 && threadIdx.x == 0) {
        float a = 0.0f;
        for (int i = 0; i < D; ++i) a += wqk[i] * wqk[i];
        *s = a;
    }
}

__global__ __launch_bounds__(256)
void k_rowdenom(const float* __restrict__ dens, const float* __restrict__ sp,
                float* __restrict__ R)
{
    __shared__ float rv[256];
    const int row = blockIdx.x, b = row / kN, tid = threadIdx.x;
    const float s = *sp, dn = dens[row];
    const float* db = dens + b * kN;
    float acc = 0.0f;
    for (int m = tid; m < kN; m += 256) acc += __expf(s * dn * db[m]);
    rv[tid] = acc; __syncthreads();
    for (int t = 128; t > 0; t >>= 1) {
        if (tid < t) rv[tid] += rv[tid + t];
        __syncthreads();
    }
    if (tid == 0) R[row] = rv[0];
}

__global__ __launch_bounds__(256)
void k_colsum(const float* __restrict__ dens, const float* __restrict__ sp,
              const float* __restrict__ R, float* __restrict__ Scol)
{
    __shared__ float rv[256];
    const int col = blockIdx.x, b = col / kN, tid = threadIdx.x;
    const float s = *sp, dm = dens[col];
    const float* db = dens + b * kN;
    const float* Rb = R + b * kN;
    float acc = 0.0f;
    for (int n = tid; n < kN; n += 256) acc += __expf(s * db[n] * dm) / Rb[n];
    rv[tid] = acc; __syncthreads();
    for (int t = 128; t > 0; t >>= 1) {
        if (tid < t) rv[tid] += rv[tid + t];
        __syncthreads();
    }
    if (tid == 0) Scol[col] = rv[0];
}

// Wt[b][m][n] = exp(s d[n] d[m]) / R[n] / (1e-9 + S[m])  (attn transposed, f16)
__global__ void k_genwt(const float* __restrict__ dens, const float* __restrict__ sp,
                        const float* __restrict__ R, const float* __restrict__ Scol,
                        f16* __restrict__ Wt)
{
    long long i = (long long)blockIdx.x * blockDim.x + threadIdx.x;
    if (i >= (long long)kB * kN * kN) return;
    const int n = (int)(i % kN);
    long long t = i / kN;
    const int m = (int)(t % kN);
    const int b = (int)(t / kN);
    const float s = *sp;
    float e = __expf(s * dens[b * kN + n] * dens[b * kN + m]);
    float w = e / R[b * kN + n] / (1e-9f + Scol[b * kN + m]);
    Wt[i] = (f16)w;
}

// ---------------------------------------------------------------------------
// Conversions / elementwise
// ---------------------------------------------------------------------------
__global__ void k_cvt(const float* __restrict__ src, f16* __restrict__ dst, long long n)
{
    long long i = (long long)blockIdx.x * blockDim.x + threadIdx.x;
    if (i < n) dst[i] = (f16)src[i];
}

// [B,C,N] f32 -> [B,N,C] f16
__global__ void k_cvt_t(const float* __restrict__ src, f16* __restrict__ dst, int C)
{
    long long i = (long long)blockIdx.x * blockDim.x + threadIdx.x;
    if (i >= (long long)kB * C * kN) return;
    const int n = (int)(i % kN);
    long long t = i / kN;
    const int c = (int)(t % C);
    const int b = (int)(t / C);
    dst[((long long)b * kN + n) * C + c] = (f16)src[i];
}

// [B,C,N]: (h - xr) f32 -> [B,N,C] f16
__global__ void k_diff_cvt_t(const float* __restrict__ h, const float* __restrict__ xr,
                             f16* __restrict__ dst, int C)
{
    long long i = (long long)blockIdx.x * blockDim.x + threadIdx.x;
    if (i >= (long long)kB * C * kN) return;
    const int n = (int)(i % kN);
    long long t = i / kN;
    const int c = (int)(t % C);
    const int b = (int)(t / C);
    dst[((long long)b * kN + n) * C + c] = (f16)(h[i] - xr[i]);
}

// conv1: 3 -> 128 (K too small for WMMA, pure VALU)
__global__ void k_conv1(const float* __restrict__ x, const float* __restrict__ w1,
                        const float* __restrict__ b1, float* __restrict__ out)
{
    long long i = (long long)blockIdx.x * blockDim.x + threadIdx.x;
    if (i >= (long long)kB * 128 * kN) return;
    const int n = (int)(i % kN);
    long long t = i / kN;
    const int o = (int)(t % 128);
    const int b = (int)(t / 128);
    const float* xr = x + ((long long)b * kN + n) * 3;
    out[i] = b1[o] + w1[o * 3] * xr[0] + w1[o * 3 + 1] * xr[1] + w1[o * 3 + 2] * xr[2];
}

// one block per channel: mean/var over (B, N)
__global__ __launch_bounds__(256)
void k_bnstats(const float* __restrict__ t, float* __restrict__ stats, int C)
{
    __shared__ float s1[256], s2[256];
    const int c = blockIdx.x, tid = threadIdx.x;
    float a = 0.0f, q = 0.0f;
    for (int j = tid; j < kB * kN; j += 256) {
        const int b = j / kN, n = j % kN;
        float v = t[((long long)b * C + c) * kN + n];
        a += v; q += v * v;
    }
    s1[tid] = a; s2[tid] = q; __syncthreads();
    for (int s = 128; s > 0; s >>= 1) {
        if (tid < s) { s1[tid] += s1[tid + s]; s2[tid] += s2[tid + s]; }
        __syncthreads();
    }
    if (tid == 0) {
        const float inv = 1.0f / (float)(kB * kN);
        float mean = s1[0] * inv;
        stats[2 * c] = mean;
        stats[2 * c + 1] = s2[0] * inv - mean * mean;
    }
}

__global__ void k_bnapply(const float* __restrict__ t, const float* __restrict__ stats,
                          float* __restrict__ h, int C, int do_relu, int residual)
{
    long long i = (long long)blockIdx.x * blockDim.x + threadIdx.x;
    if (i >= (long long)kB * C * kN) return;
    const int c = (int)((i / kN) % C);
    float y = (t[i] - stats[2 * c]) * rsqrtf(stats[2 * c + 1] + kEps);
    if (do_relu) y = fmaxf(y, 0.0f);
    h[i] = residual ? (h[i] + y) : y;
}

__global__ __launch_bounds__(256)
void k_maxpool(const float* __restrict__ h, float* __restrict__ g)
{
    __shared__ float rv[256];
    const int bc = blockIdx.x, tid = threadIdx.x;   // b*512 + c
    const float* p = h + (long long)bc * kN;
    float m = -3.0e38f;
    for (int n = tid; n < kN; n += 256) m = fmaxf(m, p[n]);
    rv[tid] = m; __syncthreads();
    for (int s = 128; s > 0; s >>= 1) {
        if (tid < s) rv[tid] = fmaxf(rv[tid], rv[tid + s]);
        __syncthreads();
    }
    if (tid == 0) g[bc] = rv[0];
}

// ---------------------------------------------------------------------------
// Head: both FC branches in a single workgroup (tiny: 8x512 input)
// ---------------------------------------------------------------------------
__device__ __forceinline__ void bnrelu_f(float* buf, int F, int tid)
{
    // BatchNorm over batch axis (8) per feature + relu, in place.
    if (tid < F) {
        float m = 0.0f;
        for (int b = 0; b < 8; ++b) m += buf[b * F + tid];
        m *= 0.125f;
        float v = 0.0f;
        for (int b = 0; b < 8; ++b) { float d = buf[b * F + tid] - m; v += d * d; }
        v *= 0.125f;
        float inv = rsqrtf(v + kEps);
        for (int b = 0; b < 8; ++b)
            buf[b * F + tid] = fmaxf((buf[b * F + tid] - m) * inv, 0.0f);
    }
}

__global__ __launch_bounds__(256)
void k_head(const float* __restrict__ g,
            const float* f1mw, const float* f1mb, const float* f2mw, const float* f2mb,
            const float* f3mw, const float* f3mb,
            const float* f1vw, const float* f1vb, const float* f2vw, const float* f2vb,
            const float* f3vw, const float* f3vb,
            float* __restrict__ out)
{
    __shared__ float t1[8 * 256];
    __shared__ float t2[8 * 256];
    const int tid = threadIdx.x;

    // ================= mean branch =================
    for (int idx = tid; idx < 8 * 256; idx += 256) {
        const int bi = idx / 256, f = idx % 256;
        float a = f1mb[f];
        const float* gw = g + bi * 512;
        const float* ww = f1mw + f * 512;
        for (int k = 0; k < 512; ++k) a += gw[k] * ww[k];
        t1[idx] = a;
    }
    __syncthreads();
    bnrelu_f(t1, 256, tid);
    __syncthreads();

    for (int idx = tid; idx < 8 * 128; idx += 256) {
        const int bi = idx / 128, f = idx % 128;
        float a = f2mb[f];
        const float* ww = f2mw + f * 256;
        for (int k = 0; k < 256; ++k) a += t1[bi * 256 + k] * ww[k];
        t2[bi * 128 + f] = a;
    }
    __syncthreads();
    bnrelu_f(t2, 128, tid);
    __syncthreads();

    for (int idx = tid; idx < 8 * 128; idx += 256) {
        const int bi = idx / 128, f = idx % 128;
        float a = f3mb[f];
        const float* ww = f3mw + f * 128;
        for (int k = 0; k < 128; ++k) a += t2[bi * 128 + k] * ww[k];
        t1[bi * 128 + f] = a;
    }
    __syncthreads();
    if (tid < 8) {                               // row-normalize u
        float s = 0.0f;
        for (int f = 0; f < 128; ++f) { float v = t1[tid * 128 + f]; s += v * v; }
        float inv = rsqrtf(s);
        for (int f = 0; f < 128; ++f) out[tid * 128 + f] = t1[tid * 128 + f] * inv;
    }
    __syncthreads();

    // ================= var branch =================
    for (int idx = tid; idx < 8 * 256; idx += 256) {
        const int bi = idx / 256, f = idx % 256;
        float a = f1vb[f];
        const float* gw = g + bi * 512;
        const float* ww = f1vw + f * 512;
        for (int k = 0; k < 512; ++k) a += gw[k] * ww[k];
        t1[idx] = a;
    }
    __syncthreads();
    bnrelu_f(t1, 256, tid);
    __syncthreads();

    for (int idx = tid; idx < 8 * 128; idx += 256) {
        const int bi = idx / 128, f = idx % 128;
        float a = f2vb[f];
        const float* ww = f2vw + f * 256;
        for (int k = 0; k < 256; ++k) a += t1[bi * 256 + k] * ww[k];
        t2[bi * 128 + f] = a;
    }
    __syncthreads();
    bnrelu_f(t2, 128, tid);
    __syncthreads();

    if (tid < 8) {
        float a = f3vb[0];
        for (int k = 0; k < 128; ++k) a += t2[tid * 128 + k] * f3vw[k];
        float sp = (a > 20.0f) ? a : __logf(1.0f + __expf(a));  // softplus
        out[1024 + tid] = sp + 1.0f;
    }
}

// ---------------------------------------------------------------------------
// Host orchestration
// ---------------------------------------------------------------------------
extern "C" void kernel_launch(void* const* d_in, const int* in_sizes, int n_in,
                              void* d_out, int out_size, void* d_ws, size_t ws_size,
                              hipStream_t stream)
{
    (void)in_sizes; (void)n_in; (void)out_size; (void)ws_size;

    const float* x      = (const float*)d_in[0];
    const float* w1     = (const float*)d_in[1];
    const float* b1     = (const float*)d_in[2];
    const float* w2     = (const float*)d_in[3];
    const float* b2     = (const float*)d_in[4];
    const float* w3     = (const float*)d_in[5];
    const float* b3     = (const float*)d_in[6];
    const float* w4     = (const float*)d_in[7];
    const float* b4     = (const float*)d_in[8];
    const float* a_wqk[4] = { (const float*)d_in[9],  (const float*)d_in[12],
                              (const float*)d_in[15], (const float*)d_in[18] };
    const float* a_wt[4]  = { (const float*)d_in[10], (const float*)d_in[13],
                              (const float*)d_in[16], (const float*)d_in[19] };
    const float* a_bt[4]  = { (const float*)d_in[11], (const float*)d_in[14],
                              (const float*)d_in[17], (const float*)d_in[20] };
    const int   a_D[4]    = { 32, 32, 64, 128 };

    char* W = (char*)d_ws;
    float* h    = (float*)(W + OFF_H);
    float* tb   = (float*)(W + OFF_T);
    f16*   hb   = (f16*)  (W + OFF_HB);
    f16*   hnc  = (f16*)  (W + OFF_HNC);
    f16*   Wt   = (f16*)  (W + OFF_WT);
    f16*   wf   = (f16*)  (W + OFF_WF);
    float* xn   = (float*)(W + OFF_XN);
    float* dinv = (float*)(W + OFF_DINV);
    float* dens = (float*)(W + OFF_DENS);
    float* bmax = (float*)(W + OFF_BMAX);
    float* sbuf = (float*)(W + OFF_S);
    float* Rrow = (float*)(W + OFF_R);
    float* Scol = (float*)(W + OFF_CS);
    float* bns  = (float*)(W + OFF_BN);
    float* g    = (float*)(W + OFF_G);

    const f16* conv_wf[3]  = { wf + WF_W2,  wf + WF_W3,  wf + WF_W4  };
    const f16* attn_wf[4]  = { wf + WF_AT1, wf + WF_AT2, wf + WF_AT3, wf + WF_AT4 };

    auto blocks1d = [](long long n) { return (unsigned)((n + 255) / 256); };

    // ---------- density ----------
    k_normalize<<<blocks1d(kB * kN), 256, 0, stream>>>(x, xn);
    k_density  <<<kB * kN, 256, 0, stream>>>(xn, dinv);
    k_batchmax <<<kB, 256, 0, stream>>>(dinv, bmax);
    k_dnorm    <<<blocks1d(kB * kN), 256, 0, stream>>>(dinv, bmax, dens);

    // ---------- weights -> f16 ----------
    k_cvt<<<blocks1d(128 * 128), 256, 0, stream>>>(w2,      wf + WF_W2,  128 * 128);
    k_cvt<<<blocks1d(256 * 128), 256, 0, stream>>>(w3,      wf + WF_W3,  256 * 128);
    k_cvt<<<blocks1d(512 * 256), 256, 0, stream>>>(w4,      wf + WF_W4,  512 * 256);
    k_cvt<<<blocks1d(128 * 128), 256, 0, stream>>>(a_wt[0], wf + WF_AT1, 128 * 128);
    k_cvt<<<blocks1d(128 * 128), 256, 0, stream>>>(a_wt[1], wf + WF_AT2, 128 * 128);
    k_cvt<<<blocks1d(256 * 256), 256, 0, stream>>>(a_wt[2], wf + WF_AT3, 256 * 256);
    k_cvt<<<blocks1d(512 * 512), 256, 0, stream>>>(a_wt[3], wf + WF_AT4, 512 * 512);

    // ---------- conv1 (3->128) + bn + relu ----------
    k_conv1  <<<blocks1d((long long)kB * 128 * kN), 256, 0, stream>>>(x, w1, b1, tb);
    k_bnstats<<<128, 256, 0, stream>>>(tb, bns, 128);
    k_bnapply<<<blocks1d((long long)kB * 128 * kN), 256, 0, stream>>>(tb, bns, h, 128, 1, 0);

    auto attn_block = [&](int C, int layer) {
        const long long elems = (long long)kB * C * kN;
        k_sumsq   <<<1, 32, 0, stream>>>(a_wqk[layer], a_D[layer], sbuf);
        k_rowdenom<<<kB * kN, 256, 0, stream>>>(dens, sbuf, Rrow);
        k_colsum  <<<kB * kN, 256, 0, stream>>>(dens, sbuf, Rrow, Scol);
        k_genwt   <<<blocks1d((long long)kB * kN * kN), 256, 0, stream>>>(dens, sbuf, Rrow, Scol, Wt);
        // x_r = h @ attn : A = f16(h) [B,C,N], Bt = Wt [B,m,n] (K=n contiguous)
        k_cvt<<<blocks1d(elems), 256, 0, stream>>>(h, hb, elems);
        k_gemm_wmma<<<dim3(C / 64, kN / 64, kB), 256, 0, stream>>>(
            hb, Wt, tb, nullptr, kN, kN,
            (long long)C * kN, (long long)kN * kN, (long long)C * kN);
        // t = wt @ (h - x_r) + bt : Bt = f16(h - xr) in [B,N,C] (K=c contiguous)
        k_diff_cvt_t<<<blocks1d(elems), 256, 0, stream>>>(h, tb, hnc, C);
        k_gemm_wmma<<<dim3(C / 64, kN / 64, kB), 256, 0, stream>>>(
            attn_wf[layer], hnc, tb, a_bt[layer], kN, C,
            0LL, (long long)kN * C, (long long)C * kN);
        // h = h + relu(bn(t))
        k_bnstats<<<C, 256, 0, stream>>>(tb, bns, C);
        k_bnapply<<<blocks1d(elems), 256, 0, stream>>>(tb, bns, h, C, 1, 1);
    };

    auto conv_block = [&](int Cin, int Cout, const f16* wfp, const float* bias, int relu) {
        k_cvt_t<<<blocks1d((long long)kB * Cin * kN), 256, 0, stream>>>(h, hnc, Cin);
        k_gemm_wmma<<<dim3(Cout / 64, kN / 64, kB), 256, 0, stream>>>(
            wfp, hnc, tb, bias, kN, Cin,
            0LL, (long long)kN * Cin, (long long)Cout * kN);
        k_bnstats<<<Cout, 256, 0, stream>>>(tb, bns, Cout);
        k_bnapply<<<blocks1d((long long)kB * Cout * kN), 256, 0, stream>>>(tb, bns, h, Cout, relu, 0);
    };

    attn_block(128, 0);
    conv_block(128, 128, conv_wf[0], b2, 1);
    attn_block(128, 1);
    conv_block(128, 256, conv_wf[1], b3, 1);
    attn_block(256, 2);
    conv_block(256, 512, conv_wf[2], b4, 0);   // conv4: no relu
    attn_block(512, 3);

    // ---------- global max pool + heads ----------
    k_maxpool<<<kB * 512, 256, 0, stream>>>(h, g);
    k_head<<<1, 256, 0, stream>>>(g,
        (const float*)d_in[21], (const float*)d_in[22],
        (const float*)d_in[23], (const float*)d_in[24],
        (const float*)d_in[25], (const float*)d_in[26],
        (const float*)d_in[27], (const float*)d_in[28],
        (const float*)d_in[29], (const float*)d_in[30],
        (const float*)d_in[31], (const float*)d_in[32],
        (float*)d_out);
}
